// MultiHeadSelfAttention_81844896793040
// MI455X (gfx1250) — matrix operand order repaired and checked by reading, not compiled, not run on hardware
//
#include <hip/hip_runtime.h>
#include <hip/hip_bf16.h>

// MHA forward for MI455X (gfx1250), wave32 + V_WMMA_F32_16X16X32_F16.
// B=2, S=2048, D=1024, H=16, DK=64.  ~69 GFLOP vs ~100MB traffic ->
// compute bound -> all matmuls via WMMA f16-in/f32-acc.
// Round 3: fully unrolled GEMM K-loops (constant trip count) so the
// scheduler pipelines loads via immediate offsets with zero register
// copies; attention loop unrolled x2 for cross-iteration overlap.

typedef __attribute__((ext_vector_type(16))) _Float16 v16h;
typedef __attribute__((ext_vector_type(8)))  float    v8f;

#define BB 2
#define SS 2048
#define DD 1024
#define HH 16
#define DKK 64
#define NQKV 3072

__device__ __forceinline__ v8f wmma_f16(v16h a, v16h b, v8f c) {
  return __builtin_amdgcn_wmma_f32_16x16x32_f16(
      false, a, false, b, (short)0, c, false, false);
}

// A-matrix 16x32 f16 fragment (ISA 7.12.2): lane L, M = L%16, g = L/16.
// halves 0..7  -> K = c0 + g*8 + i ; halves 8..15 -> K = c0 + 16 + g*8 + i-8.
__device__ __forceinline__ v16h load_fragA(const _Float16* rowp, int c0, int g) {
  const unsigned* p = (const unsigned*)rowp;
  union { v16h h; unsigned u[8]; } f;
  const int b0 = (c0 + g * 8) >> 1;
  const int b1 = (c0 + 16 + g * 8) >> 1;
#pragma unroll
  for (int q = 0; q < 4; ++q) f.u[q] = p[b0 + q];
#pragma unroll
  for (int q = 0; q < 4; ++q) f.u[4 + q] = p[b1 + q];
  return f.h;
}

// B-matrix 32x16 f16 fragment: lane L, N = L%16, g = L/16,
// halves 0..7 -> K = c0 + g*8 + i ; halves 8..15 -> K = c0 + 16 + g*8 + i-8.
__device__ __forceinline__ v16h load_fragB(const _Float16* rowp, int c0, int g) {
  const unsigned* p = (const unsigned*)rowp;
  union { v16h h; unsigned u[8]; } f;
  const int b0 = (c0 + g * 8) >> 1;
  const int b1 = (c0 + 16 + g * 8) >> 1;
#pragma unroll
  for (int q = 0; q < 4; ++q) f.u[q] = p[b0 + q];
#pragma unroll
  for (int q = 0; q < 4; ++q) f.u[4 + q] = p[b1 + q];
  return f.h;
}

__device__ __forceinline__ float redmax16(float v) {
#pragma unroll
  for (int m = 1; m < 16; m <<= 1) v = fmaxf(v, __shfl_xor(v, m, 32));
  return v;
}
__device__ __forceinline__ float redsum16(float v) {
#pragma unroll
  for (int m = 1; m < 16; m <<= 1) v += __shfl_xor(v, m, 32);
  return v;
}

// ---------------- precision / layout prep ----------------
__global__ void k_f32_to_f16(const float* __restrict__ src,
                             _Float16* __restrict__ dst, int n) {
  int i = blockIdx.x * blockDim.x + threadIdx.x;
  if (i < n) dst[i] = (_Float16)src[i];
}

// dst[n*K + k] = (f16) src[k*N + n]
__global__ void k_transpose_f16(const float* __restrict__ src,
                                _Float16* __restrict__ dst, int K, int N) {
  int i = blockIdx.x * blockDim.x + threadIdx.x;
  if (i >= K * N) return;
  int n = i / K, k = i % K;
  dst[i] = (_Float16)src[(size_t)k * N + n];
}

// ---------------- QKV projection GEMM ----------------
// 32x64 tile per wave; K loop fully unrolled (32 steps, immediate offsets).
// grid (128, 48).  V written TRANSPOSED: Vt[b][h][dk][s].
__global__ __launch_bounds__(32) void k_gemm_qkv(
    const _Float16* __restrict__ xh, const _Float16* __restrict__ wT,
    const float* __restrict__ bias, _Float16* __restrict__ Qh,
    _Float16* __restrict__ Kh, _Float16* __restrict__ Vt) {
  const int lane = threadIdx.x, g = lane >> 4, ln = lane & 15;
  const int mbase = blockIdx.x * 32, nbase = blockIdx.y * 64;
  v8f acc[2][4] = {};
  const _Float16* arow0 = xh + (size_t)(mbase + ln) * DD;
  const _Float16* arow1 = xh + (size_t)(mbase + 16 + ln) * DD;
  const _Float16* wrow[4];
#pragma unroll
  for (int t = 0; t < 4; ++t) wrow[t] = wT + (size_t)(nbase + t * 16 + ln) * DD;

#pragma unroll
  for (int kb = 0; kb < DD; kb += 32) {
    const v16h a0 = load_fragA(arow0, kb, g);
    const v16h a1 = load_fragA(arow1, kb, g);
#pragma unroll
    for (int t = 0; t < 4; ++t) {
      const v16h bf = load_fragB(wrow[t], kb, g);
      acc[0][t] = wmma_f16(a0, bf, acc[0][t]);
      acc[1][t] = wmma_f16(a1, bf, acc[1][t]);
    }
  }

#pragma unroll
  for (int t = 0; t < 4; ++t) {
    const int n = nbase + t * 16 + ln;
    const float bv = bias[n];
    const int h = n / 192, r = n % 192;
    const int sel = r >> 6, dk = r & 63;
#pragma unroll
    for (int rs = 0; rs < 2; ++rs) {
#pragma unroll
      for (int j = 0; j < 8; ++j) {
        const int tok = mbase + rs * 16 + j + 8 * g;
        const int b = tok >> 11, s = tok & (SS - 1);
        const _Float16 v = (_Float16)(acc[rs][t][j] + bv);
        if (sel == 0)
          Qh[(((size_t)(b * HH + h) * SS + s) << 6) + dk] = v;
        else if (sel == 1)
          Kh[(((size_t)(b * HH + h) * SS + s) << 6) + dk] = v;
        else
          Vt[((size_t)(b * HH + h) * DKK + dk) * SS + s] = v;  // transposed
      }
    }
  }
}

// ---------------- flash attention ----------------
// grid (S/16, B*H), 1 wave/block, online softmax over 64 key steps of 32.
__global__ __launch_bounds__(32) void k_attn(
    const _Float16* __restrict__ Qh, const _Float16* __restrict__ Kh,
    const _Float16* __restrict__ Vt, const int* __restrict__ mask,
    _Float16* __restrict__ Oh) {
  __shared__ __align__(16) _Float16 sP[16 * 32];
  const int lane = threadIdx.x, g = lane >> 4, ln = lane & 15;
  const int qbase = blockIdx.x * 16;
  const int bh = blockIdx.y, b = bh >> 4;
  const _Float16* Qp = Qh + (size_t)bh * SS * DKK;
  const _Float16* Kp = Kh + (size_t)bh * SS * DKK;
  const _Float16* Vp = Vt + (size_t)bh * DKK * SS;  // [dk][s]
  const int* mp = mask + b * SS;

  const v16h qf0 = load_fragA(Qp + (size_t)(qbase + ln) * DKK, 0, g);
  const v16h qf1 = load_fragA(Qp + (size_t)(qbase + ln) * DKK, 32, g);
  const _Float16* vrow[4];
#pragma unroll
  for (int t = 0; t < 4; ++t) vrow[t] = Vp + (size_t)(t * 16 + ln) * SS;

  v8f acc[4] = {};
  float mrow[8], lrow[8];
#pragma unroll
  for (int j = 0; j < 8; ++j) { mrow[j] = -1.0e30f; lrow[j] = 0.0f; }

  // scores scaled into log2 domain: softmax(exp2(s')) with s' = s*scale*log2e
  const float SC = 0.125f * 1.44269504088896f;

#pragma unroll 2
  for (int k0 = 0; k0 < SS; k0 += 32) {
    // ---- issue ALL loads for this step up front ----
    const int mv0 = mp[k0 + ln];
    const int mv1 = mp[k0 + 16 + ln];
    const _Float16* krow0 = Kp + (size_t)(k0 + ln) * DKK;
    const _Float16* krow1 = Kp + (size_t)(k0 + 16 + ln) * DKK;
    v16h kf[4], vf[4];
    kf[0] = load_fragB(krow0, 0, g);
    kf[1] = load_fragB(krow0, 32, g);
    kf[2] = load_fragB(krow1, 0, g);
    kf[3] = load_fragB(krow1, 32, g);
#pragma unroll
    for (int t = 0; t < 4; ++t) vf[t] = load_fragB(vrow[t], k0, g);
    // prefetch next key block's K/V cachelines (global_prefetch_b8)
    __builtin_prefetch(Kp + (size_t)(k0 + 32 + ln) * DKK, 0, 1);
    __builtin_prefetch(Kp + (size_t)(k0 + 48 + ln) * DKK, 0, 1);
    __builtin_prefetch(vrow[lane & 3] + k0 + 32, 0, 1);

    // ---- scores ----
    v8f sc0 = {}, sc1 = {};
    sc0 = wmma_f16(qf0, kf[0], sc0);
    sc0 = wmma_f16(qf1, kf[1], sc0);
    sc1 = wmma_f16(qf0, kf[2], sc1);
    sc1 = wmma_f16(qf1, kf[3], sc1);

    // ---- online softmax (V loads still in flight) ----
    float p0[8], p1[8];
#pragma unroll
    for (int j = 0; j < 8; ++j) {
      float s0 = sc0[j] * SC;
      float s1 = sc1[j] * SC;
      if (mv0 == 0) s0 = -1.0e30f;  // finite sentinel self-corrects
      if (mv1 == 0) s1 = -1.0e30f;
      const float tmax = redmax16(fmaxf(s0, s1));
      const float mn = fmaxf(mrow[j], tmax);
      const float r = exp2f(mrow[j] - mn);
      const float e0 = exp2f(s0 - mn);
      const float e1 = exp2f(s1 - mn);
      lrow[j] = lrow[j] * r + redsum16(e0 + e1);
      mrow[j] = mn;
      p0[j] = e0; p1[j] = e1;
#pragma unroll
      for (int t = 0; t < 4; ++t) acc[t][j] *= r;
    }
    // C-layout P -> A-layout via LDS (single wave, DS in-order)
#pragma unroll
    for (int j = 0; j < 8; ++j) {
      const int row = j + 8 * g;
      sP[row * 32 + ln]      = (_Float16)p0[j];
      sP[row * 32 + 16 + ln] = (_Float16)p1[j];
    }
    const v16h pa = load_fragA(sP + ln * 32, 0, g);
#pragma unroll
    for (int t = 0; t < 4; ++t) acc[t] = wmma_f16(pa, vf[t], acc[t]);
  }

#pragma unroll
  for (int j = 0; j < 8; ++j) {
    const float inv = (lrow[j] > 0.0f) ? 1.0f / lrow[j] : 0.0f;
#pragma unroll
    for (int t = 0; t < 4; ++t)
      Oh[(size_t)(bh * SS + qbase + j + 8 * g) * DKK + t * 16 + ln] =
          (_Float16)(acc[t][j] * inv);
  }
}

// ---------------- output projection GEMM ----------------
// 32x64 tile per wave; K loop fully unrolled.  grid (128, 16).
__global__ __launch_bounds__(32) void k_gemm_out(
    const _Float16* __restrict__ Oh, const _Float16* __restrict__ wT,
    const float* __restrict__ bias, const float* __restrict__ gamma,
    float* __restrict__ out) {
  const int lane = threadIdx.x, g = lane >> 4, ln = lane & 15;
  const int mbase = blockIdx.x * 32, nbase = blockIdx.y * 64;
  const int tokA0 = mbase + ln,      bA0 = tokA0 >> 11, sA0 = tokA0 & (SS - 1);
  const int tokA1 = mbase + 16 + ln, bA1 = tokA1 >> 11, sA1 = tokA1 & (SS - 1);
  const _Float16* arow0 = Oh + ((size_t)(bA0 * HH) * SS + sA0) * DKK;
  const _Float16* arow1 = Oh + ((size_t)(bA1 * HH) * SS + sA1) * DKK;
  const _Float16* wrow[4];
#pragma unroll
  for (int t = 0; t < 4; ++t) wrow[t] = wT + (size_t)(nbase + t * 16 + ln) * DD;
  v8f acc[2][4] = {};

#pragma unroll
  for (int kb = 0; kb < DD; kb += 32) {
    // A row for head h = kb/64; 32-wide frag never crosses a head boundary.
    const int h = kb >> 6, c0m = kb & 63;
    const size_t hoff = (size_t)h * SS * DKK;
    const v16h a0 = load_fragA(arow0 + hoff, c0m, g);
    const v16h a1 = load_fragA(arow1 + hoff, c0m, g);
#pragma unroll
    for (int t = 0; t < 4; ++t) {
      const v16h bf = load_fragB(wrow[t], kb, g);
      acc[0][t] = wmma_f16(a0, bf, acc[0][t]);
      acc[1][t] = wmma_f16(a1, bf, acc[1][t]);
    }
  }

#pragma unroll
  for (int t = 0; t < 4; ++t) {
    const int n = nbase + t * 16 + ln;
    const float bv = bias[n];
    const float gm = gamma[n] + 1.0f;
#pragma unroll
    for (int rs = 0; rs < 2; ++rs) {
#pragma unroll
      for (int j = 0; j < 8; ++j) {
        const int tok = mbase + rs * 16 + j + 8 * g;
        out[(size_t)tok * DD + n] = (acc[rs][t][j] + bv) * gm;
      }
    }
  }
}

extern "C" void kernel_launch(void* const* d_in, const int* in_sizes, int n_in,
                              void* d_out, int out_size, void* d_ws, size_t ws_size,
                              hipStream_t stream) {
  const float* x      = (const float*)d_in[0];
  const float* gamma  = (const float*)d_in[1];
  const int*   mask   = (const int*)d_in[2];
  const float* wqkv_w = (const float*)d_in[3];
  const float* wqkv_b = (const float*)d_in[4];
  const float* wo_w   = (const float*)d_in[5];
  const float* wo_b   = (const float*)d_in[6];
  float* out = (float*)d_out;

  char* ws = (char*)d_ws;
  size_t off = 0;
  auto carve = [&](size_t bytes) {
    char* p = ws + off;
    off = (off + bytes + 255) & ~(size_t)255;
    return p;
  };
  const size_t nX  = (size_t)BB * SS * DD;
  const size_t nWq = (size_t)DD * NQKV;
  const size_t nWo = (size_t)DD * DD;
  const size_t nHd = (size_t)BB * HH * SS * DKK;

  _Float16* xh    = (_Float16*)carve(nX * 2);
  _Float16* wqkvT = (_Float16*)carve(nWq * 2);
  _Float16* woT   = (_Float16*)carve(nWo * 2);
  _Float16* Qh    = (_Float16*)carve(nHd * 2);
  _Float16* Kh    = (_Float16*)carve(nHd * 2);
  _Float16* Vt    = (_Float16*)carve(nHd * 2);
  _Float16* Oh    = (_Float16*)carve(nHd * 2);

  k_f32_to_f16<<<dim3((unsigned)((nX + 255) / 256)), 256, 0, stream>>>(x, xh, (int)nX);
  k_transpose_f16<<<dim3((unsigned)((nWq + 255) / 256)), 256, 0, stream>>>(wqkv_w, wqkvT, DD, NQKV);
  k_transpose_f16<<<dim3((unsigned)((nWo + 255) / 256)), 256, 0, stream>>>(wo_w, woT, DD, DD);

  k_gemm_qkv<<<dim3(BB * SS / 32, NQKV / 64), 32, 0, stream>>>(
      xh, wqkvT, wqkv_b, Qh, Kh, Vt);
  k_attn<<<dim3(SS / 16, BB * HH), 32, 0, stream>>>(Qh, Kh, Vt, mask, Oh);
  k_gemm_out<<<dim3(BB * SS / 32, DD / 64), 32, 0, stream>>>(
      Oh, woT, wo_b, gamma, out);
}
